// SECAM_Module_37434934952404
// MI455X (gfx1250) — matrix operand order repaired and checked by reading, not compiled, not run on hardware
//
#include <hip/hip_runtime.h>
#include <hip/hip_bf16.h>

// Problem constants (from reference): x (8, 64, 256, 256) f32
#define BATCH 8
#define CCH   64
#define HW    65536          // 256*256
#define LSTR  72             // LDS row stride (ushorts): 144B rows, 16B aligned

typedef __attribute__((ext_vector_type(16))) __bf16 v16bf;
typedef __attribute__((ext_vector_type(8)))  __bf16 v8bf;
typedef __attribute__((ext_vector_type(8)))  float  v8f;

union BF16x16 {
    unsigned int u[8];
    v16bf v;
};

// Convert 8 consecutive f32 -> 8 bf16 (4x v_cvt_pk_bf16_f32), one 16B LDS store.
__device__ __forceinline__ void cvt_store8(unsigned short* lds, v8f f) {
    *(v8bf*)lds = __builtin_convertvector(f, v8bf);
}

// ---------------------------------------------------------------------------
// Kernel 0: zero the Gram accumulator (atomics target)
// ---------------------------------------------------------------------------
__global__ void zero_energy(float* __restrict__ e) {
    int i = blockIdx.x * blockDim.x + threadIdx.x;
    if (i < BATCH * CCH * CCH) e[i] = 0.0f;
}

// ---------------------------------------------------------------------------
// Kernel 1: Gram matrix energy[b,c,d] = sum_n x[b,c,n] * x[b,d,n]
// grid = (64 N-splits, 8 batches), 512 threads = 16 waves.
// Wave w owns 16x16 tile (c0=(w>>2)*16, d0=(w&3)*16); accumulates over its
// 1024-wide N range in chunks of 64 (two K=32 bf16 WMMAs per chunk).
// Thread t loads 8 consecutive floats (2x global_load_b128 -> ds_store_b128).
// ---------------------------------------------------------------------------
__global__ __launch_bounds__(512)
void gram_kernel(const float* __restrict__ x, float* __restrict__ energy) {
    __shared__ unsigned short xs[CCH * LSTR];   // [channel][spatial] bf16

    const int tid  = threadIdx.x;
    const int lane = tid & 31;
    const int wave = tid >> 5;            // 0..15
    const int c0   = (wave >> 2) * 16;    // A rows (channels c)
    const int d0   = (wave & 3)  * 16;    // B cols (channels d)
    const int b    = blockIdx.y;

    // Static load assignment: row = tid>>3 (0..63), col = (tid&7)*8
    const int lr = tid >> 3;
    const int lc = (tid & 7) * 8;
    const float* srcbase = x + (size_t)(b * CCH + lr) * HW + lc;
    unsigned short* ldst = &xs[lr * LSTR + lc];

    v8f acc = {};

    for (int it = 0; it < 16; ++it) {
        const int nchunk = blockIdx.x * 1024 + it * 64;
        const float* src = srcbase + nchunk;

        v8f f = *(const v8f*)(src);              // 32B (2x global_load_b128)
        if (it + 1 < 16) __builtin_prefetch(src + 64, 0, 1);
        cvt_store8(ldst, f);
        __syncthreads();

        #pragma unroll
        for (int kh = 0; kh < 2; ++kh) {
            const int kbase = kh * 32;
            BF16x16 A, Bm;
            #pragma unroll
            for (int i = 0; i < 8; ++i) {
                // A (16x32 bf16): lane m=lane&15; VGPR i holds K pair per ISA table
                int ka = kbase + ((i < 4) ? 2 * i : 16 + 2 * (i - 4))
                               + ((lane >= 16) ? 8 : 0);
                int m  = lane & 15;
                A.u[i] = *(const unsigned int*)&xs[(c0 + m) * LSTR + ka];
                // B (32x16 bf16): same K order as A (VGPR i holds K pair per ISA table)
                int kb = kbase + ((i < 4) ? 2 * i : 16 + 2 * (i - 4))
                               + ((lane >= 16) ? 8 : 0);
                int nc = lane & 15;
                Bm.u[i] = *(const unsigned int*)&xs[(d0 + nc) * LSTR + kb];
            }
            acc = __builtin_amdgcn_wmma_f32_16x16x32_bf16(
                false, A.v, false, Bm.v, (short)0, acc, false, false);
        }
        __syncthreads();
    }

    // Flush 16x16 f32 partial tile (C/D layout: VGPR i -> rows i and i+8)
    #pragma unroll
    for (int i = 0; i < 8; ++i) {
        int c = c0 + i + ((lane >= 16) ? 8 : 0);
        int d = d0 + (lane & 15);
        atomicAdd(&energy[(b * CCH + c) * CCH + d], acc[i]);
    }
}

// ---------------------------------------------------------------------------
// Kernel 2: diag-rsqrt normalize -> SE bottleneck (4096->16->4096) -> softmax.
// One 256-thread block per batch; everything fits in LDS.
// ---------------------------------------------------------------------------
__global__ __launch_bounds__(256)
void se_softmax_kernel(const float* __restrict__ energy,
                       const float* __restrict__ W1, const float* __restrict__ b1,
                       const float* __restrict__ W2, const float* __restrict__ b2,
                       float* __restrict__ att) {
    __shared__ float e[CCH * CCH];   // 16 KB
    __shared__ float r[CCH];
    __shared__ float h[16];
    __shared__ float red[256];

    const int tid = threadIdx.x;
    const int b   = blockIdx.x;
    const float* eb = energy + b * CCH * CCH;

    for (int i = tid; i < CCH * CCH; i += 256) e[i] = eb[i];
    __syncthreads();

    if (tid < CCH) r[tid] = rsqrtf(fmaxf(e[tid * CCH + tid], 1e-12f));
    __syncthreads();

    for (int i = tid; i < CCH * CCH; i += 256) e[i] *= r[i >> 6] * r[i & 63];
    __syncthreads();

    // h = relu(W1 @ e + b1), 16 outputs via LDS tree reduction
    for (int o = 0; o < 16; ++o) {
        float p = 0.0f;
        const float* w = W1 + o * (CCH * CCH);
        for (int i = tid; i < CCH * CCH; i += 256) p += e[i] * w[i];
        red[tid] = p;
        __syncthreads();
        for (int s = 128; s > 0; s >>= 1) {
            if (tid < s) red[tid] += red[tid + s];
            __syncthreads();
        }
        if (tid == 0) h[o] = fmaxf(red[0] + b1[o], 0.0f);
        __syncthreads();
    }

    // e2 = W2 @ h + b2 (overwrite e)
    for (int i = tid; i < CCH * CCH; i += 256) {
        float v = b2[i];
        const float* w = W2 + i * 16;
        #pragma unroll
        for (int o = 0; o < 16; ++o) v += w[o] * h[o];
        e[i] = v;
    }
    __syncthreads();

    // softmax(rowmax - e) over last dim == exp(rowmin - e) normalized
    if (tid < CCH) {
        const float* row = e + tid * CCH;
        float mn = row[0];
        for (int d = 1; d < CCH; ++d) mn = fminf(mn, row[d]);
        float sum = 0.0f;
        float* arow = att + (b * CCH + tid) * CCH;
        for (int d = 0; d < CCH; ++d) {
            float p = __expf(mn - row[d]);
            arow[d] = p;
            sum += p;
        }
        float inv = 1.0f / sum;
        for (int d = 0; d < CCH; ++d) arow[d] *= inv;
    }
}

// ---------------------------------------------------------------------------
// Kernel 3: out = gamma * (att @ xf) + x.
// grid = (1024 N-tiles, 8 batches), 512 threads = 16 waves; wave computes a
// 16(M=c) x 16(N=spatial) tile with K=64 (two bf16 WMMAs).
// ---------------------------------------------------------------------------
__global__ __launch_bounds__(512)
void apply_kernel(const float* __restrict__ x, const float* __restrict__ att,
                  const float* __restrict__ gamma, float* __restrict__ out) {
    __shared__ unsigned short as_[CCH * LSTR];   // attention bf16 [c][d]
    __shared__ unsigned short xsb[CCH * LSTR];   // x chunk bf16 [d][n]

    const int tid  = threadIdx.x;
    const int lane = tid & 31;
    const int wave = tid >> 5;
    const int c0   = (wave >> 2) * 16;
    const int n1   = (wave & 3)  * 16;
    const int b    = blockIdx.y;
    const int n0   = blockIdx.x * 64;
    const float g  = gamma[0];

    // Static load assignment: row = tid>>3 (0..63), col = (tid&7)*8
    const int lr = tid >> 3;
    const int lc = (tid & 7) * 8;

    {
        v8f t = *(const v8f*)(att + (size_t)(b * CCH + lr) * CCH + lc);
        v8f u = *(const v8f*)(x + (size_t)(b * CCH + lr) * HW + n0 + lc);
        cvt_store8(&as_[lr * LSTR + lc], t);
        cvt_store8(&xsb[lr * LSTR + lc], u);
    }
    __syncthreads();

    v8f acc = {};
    #pragma unroll
    for (int kh = 0; kh < 2; ++kh) {
        const int kbase = kh * 32;
        BF16x16 A, Bm;
        #pragma unroll
        for (int i = 0; i < 8; ++i) {
            int ka = kbase + ((i < 4) ? 2 * i : 16 + 2 * (i - 4))
                           + ((lane >= 16) ? 8 : 0);
            int m  = lane & 15;
            A.u[i] = *(const unsigned int*)&as_[(c0 + m) * LSTR + ka];

            int kb = kbase + ((i < 4) ? 2 * i : 16 + 2 * (i - 4))
                           + ((lane >= 16) ? 8 : 0);
            int nn = n1 + (lane & 15);
            unsigned int lo = xsb[kb * LSTR + nn];
            unsigned int hi = xsb[(kb + 1) * LSTR + nn];
            Bm.u[i] = lo | (hi << 16);
        }
        acc = __builtin_amdgcn_wmma_f32_16x16x32_bf16(
            false, A.v, false, Bm.v, (short)0, acc, false, false);
    }

    // Residual + gamma scale, write per C/D layout (VGPR i -> rows i, i+8)
    #pragma unroll
    for (int i = 0; i < 8; ++i) {
        int row = c0 + i + ((lane >= 16) ? 8 : 0);
        int col = n0 + n1 + (lane & 15);
        size_t gidx = (size_t)(b * CCH + row) * HW + col;
        out[gidx] = g * acc[i] + x[gidx];
    }
}

// ---------------------------------------------------------------------------
extern "C" void kernel_launch(void* const* d_in, const int* in_sizes, int n_in,
                              void* d_out, int out_size, void* d_ws, size_t ws_size,
                              hipStream_t stream) {
    const float* x     = (const float*)d_in[0];
    const float* W1    = (const float*)d_in[1];
    const float* b1    = (const float*)d_in[2];
    const float* W2    = (const float*)d_in[3];
    const float* b2    = (const float*)d_in[4];
    const float* gamma = (const float*)d_in[5];
    float* out = (float*)d_out;

    float* energy = (float*)d_ws;                 // 8*64*64 f32
    float* att    = energy + BATCH * CCH * CCH;   // 8*64*64 f32

    zero_energy<<<(BATCH * CCH * CCH + 255) / 256, 256, 0, stream>>>(energy);
    gram_kernel<<<dim3(64, BATCH), 512, 0, stream>>>(x, energy);
    se_softmax_kernel<<<BATCH, 256, 0, stream>>>(energy, W1, b1, W2, b2, att);
    apply_kernel<<<dim3(HW / 64, BATCH), 512, 0, stream>>>(x, att, gamma, out);
}
